// OlmoLMHeadCE_49873160241202
// MI455X (gfx1250) — matrix operand order repaired and checked by reading, not compiled, not run on hardware
//
#include <hip/hip_runtime.h>
#include <hip/hip_bf16.h>

// ---------------------------------------------------------------------------
// Fused LM-head + cross-entropy (mean, ignore_index=-100) for gfx1250.
//   Pass 1: x[4096,2048] f32 -> xb bf16            (ws)
//   Pass 2: W[2048,50257] f32 -> wT[50304,2048] bf16, transposed + zero-pad (ws)
//   Pass 3: streaming GEMM via v_wmma_f32_16x16x32_bf16 with the weight
//           stream DMA'd into LDS by the Tensor Data Mover (double-buffered),
//           online logsumexp per row, single scalar loss out.
// Workspace layout: [0,8)   loss/count accumulators
//                   [256, +16MB)  xb
//                   [16MB+256, +197MB) wT          (needs ws_size >= ~213MB)
// ---------------------------------------------------------------------------

typedef __attribute__((ext_vector_type(16))) __bf16        v16bf;
typedef __attribute__((ext_vector_type(8)))  float         v8f;
typedef __attribute__((ext_vector_type(4)))  unsigned int  v4u;
typedef __attribute__((ext_vector_type(4)))  unsigned int  u32x4;
typedef __attribute__((ext_vector_type(8)))  int           i32x8;
typedef __attribute__((ext_vector_type(4)))  int           i32x4;

#define N_ROWS     4096
#define H_DIM      2048
#define V_DIM      50257
#define V_PAD      50304                // 393 * 128, zero-padded in wT
#define M_TILE     32                   // rows per workgroup
#define V_STEP     128                  // cols per super-tile (8 waves * 16)
#define K_CHUNK    256                  // K elems per TDM chunk
#define XS_STRIDE  2056                 // 2048 + 8 pad (ushorts)
#define BS_STRIDE  264                  // 256 + 8 pad (ushorts) == TDM pad layout
#define XS_BYTES   (M_TILE * XS_STRIDE * 2)     // 131584
#define STAT_BYTES ((8 * M_TILE * 2 + M_TILE) * 4)
#define BS0_OFF    (XS_BYTES + STAT_BYTES)      // 133760 (16B aligned)
#define BS_BYTES   (V_STEP * BS_STRIDE * 2)     // 67584
#define BS1_OFF    (BS0_OFF + BS_BYTES)
#define SMEM_TOTAL (BS1_OFF + BS_BYTES)         // 268928 < 320KB
#define NCHUNKS    ((V_PAD / V_STEP) * (H_DIM / K_CHUNK))  // 393*8 = 3144
#define IGNORE_IDX (-100)
#define NEG_BIG    (-3.0e38f)

#define WS_XB_OFF  256
#define WS_WT_OFF  (256 + (size_t)N_ROWS * H_DIM * 2)

#if __has_builtin(__builtin_amdgcn_tensor_load_to_lds)
#define USE_TDM 1
#endif

__device__ __forceinline__ unsigned short f32_to_bf16_rne(float f) {
    unsigned int u = __builtin_bit_cast(unsigned int, f);
    unsigned int r = u + 0x7FFFu + ((u >> 16) & 1u);
    return (unsigned short)(r >> 16);
}

__device__ __forceinline__ void lse_merge(float& m, float& s, float om, float os) {
    float nm = fmaxf(m, om);
    s = s * __expf(m - nm) + os * __expf(om - nm);
    m = nm;
}

#ifdef USE_TDM
// Issue a 2D bf16 TDM load: tile [V_STEP rows][K_CHUNK contiguous elems],
// row stride = H_DIM elems, LDS-padded by 4 dwords every 128 dwords.
// clang-23 builtin arity: (g0 u32x4, g1 i32x8, g2 i32x4, g3 i32x4, g4 i32x8, cpol)
__device__ __forceinline__ void tdm_load_2d(unsigned lds_off,
                                            unsigned long long gaddr) {
    u32x4 g0;
    g0[0] = 1u;                                   // count=1 (valid user D#)
    g0[1] = lds_off;                              // lds_addr (bytes)
    g0[2] = (unsigned)(gaddr & 0xFFFFFFFFull);    // global_addr[31:0]
    g0[3] = (unsigned)((gaddr >> 32) & 0x1FFFFFFull) | (2u << 30); // [56:32], type=2
    i32x8 g1;
    // data_size=2B (code 1), pad_enable, pad_interval=128dw (code 6),
    // pad_amount=4dw (code 3); workgroup_mask=0 (not clustered)
    g1[0] = (int)((1u << 16) | (1u << 20) | (6u << 22) | (3u << 25));
    g1[1] = (int)((H_DIM & 0xFFFFu) << 16);                        // tensor_dim0 lo
    g1[2] = (int)(((H_DIM >> 16) & 0xFFFFu) | ((V_PAD & 0xFFFFu) << 16));
    g1[3] = (int)(((V_PAD >> 16) & 0xFFFFu) | ((unsigned)K_CHUNK << 16)); // tile_dim0
    g1[4] = (int)(V_STEP & 0xFFFFu);                               // tile_dim1, tile_dim2=0
    g1[5] = (int)(unsigned)H_DIM;                                  // stride0[31:0]
    g1[6] = 0;                                                     // stride0[47:32], stride1 lo
    g1[7] = 0;
    i32x4 z4 = {0, 0, 0, 0};
    i32x8 z8 = {0, 0, 0, 0, 0, 0, 0, 0};
    __builtin_amdgcn_tensor_load_to_lds(g0, g1, z4, z4, z8, 0);
}
#endif

__global__ __launch_bounds__(256) void OlmoLMHeadCE_main(
    const unsigned short* __restrict__ xb, const unsigned short* __restrict__ wT,
    const int* __restrict__ tgt, float* __restrict__ ws) {
    extern __shared__ char smem[];
    unsigned short* xs   = (unsigned short*)smem;                  // [32][2056] bf16
    float*          stM  = (float*)(smem + XS_BYTES);              // [8][32]
    float*          stS  = stM + 8 * M_TILE;
    float*          tgtL = stS + 8 * M_TILE;                       // [32]

    const int tid   = threadIdx.x;
    const int lane  = tid & 31;
    const int wv    = tid >> 5;
    const int n     = lane & 15;
    const int hi    = lane >> 4;
    const int mbase = blockIdx.x * M_TILE;
    const int myColOff = wv * 16 + n;

    // ---- stage pre-converted x tile into LDS (vector copies, once) -----
    #pragma unroll 4
    for (int i = 0; i < 32; ++i) {
        int idx = (i * 256 + tid) * 8;        // 32*2048 bf16, 8 at a time
        int r = idx >> 11, c = idx & 2047;
        *(v4u*)(xs + r * XS_STRIDE + c) =
            *(const v4u*)(xb + (size_t)(mbase + r) * H_DIM + c);
    }
    if (tid < M_TILE) tgtL[tid] = 0.0f;

    int tg0[8], tg1[8];
    #pragma unroll
    for (int r = 0; r < 8; ++r) {
        tg0[r] = tgt[mbase + r + 8 * hi];
        tg1[r] = tgt[mbase + 16 + r + 8 * hi];
    }
    float mx0[8], sm0[8], mx1[8], sm1[8];
    #pragma unroll
    for (int r = 0; r < 8; ++r) { mx0[r] = NEG_BIG; sm0[r] = 0.f; mx1[r] = NEG_BIG; sm1[r] = 0.f; }

    const unsigned long long wtBase = (unsigned long long)(uintptr_t)wT;

#ifdef USE_TDM
    if (wv == 0)   // prologue DMA: chunk 0 -> buffer 0
        tdm_load_2d(BS0_OFF, wtBase + 0ull);
#endif
    __syncthreads();

    v8f c0 = {};
    v8f c1 = {};

    #pragma unroll 1
    for (int it = 0; it < NCHUNKS; ++it) {
        const int nb     = it & 1;
        const int chunk  = it & 7;
        const int vbase  = (it >> 3) * V_STEP;
        unsigned short* bs =
            (unsigned short*)(smem + (nb ? BS1_OFF : BS0_OFF));

#ifdef USE_TDM
        if (wv == 0) {
            if (it + 1 < NCHUNKS) {
                const int nit = it + 1;
                const unsigned long long ng = wtBase +
                    (((unsigned long long)(nit >> 3) * V_STEP) * H_DIM +
                     (unsigned long long)(nit & 7) * K_CHUNK) * 2ull;
                tdm_load_2d(nb ? BS0_OFF : BS1_OFF, ng);
                __builtin_amdgcn_s_wait_tensorcnt(1);  // current buffer complete
            } else {
                __builtin_amdgcn_s_wait_tensorcnt(0);
            }
        }
#else
        { // fallback: cooperative bf16 copy with the same padded layout
            const unsigned short* src =
                wT + (size_t)vbase * H_DIM + (size_t)chunk * K_CHUNK;
            #pragma unroll 4
            for (int i = 0; i < 16; ++i) {
                int idx = (i * 256 + tid) * 8;       // 128*256 elems
                int row = idx >> 8, k = idx & 255;
                *(v4u*)(bs + row * BS_STRIDE + k) =
                    *(const v4u*)(src + (size_t)row * H_DIM + k);
            }
        }
#endif
        __syncthreads();

        const int kchunk = chunk * K_CHUNK;

        #pragma unroll
        for (int kk = 0; kk < K_CHUNK; kk += 32) {
            union Frag { v4u u[2]; v16bf v; };
            Frag a0, a1, b;
            // A: lane holds row n (sub0) / n+16 (sub1); K split {0-7,16-23}/{8-15,24-31}
            const unsigned short* pa0 = xs + n * XS_STRIDE + kchunk + kk + 8 * hi;
            a0.u[0] = *(const v4u*)pa0;
            a0.u[1] = *(const v4u*)(pa0 + 16);
            const unsigned short* pa1 = pa0 + 16 * XS_STRIDE;
            a1.u[0] = *(const v4u*)pa1;
            a1.u[1] = *(const v4u*)(pa1 + 16);
            // B: lane holds col n; K split {0-7,16-23}/{8-15,24-31} (matches A)
            const unsigned short* pb = bs + myColOff * BS_STRIDE + kk + 8 * hi;
            b.u[0] = *(const v4u*)pb;
            b.u[1] = *(const v4u*)(pb + 16);

            c0 = __builtin_amdgcn_wmma_f32_16x16x32_bf16(false, a0.v, false, b.v,
                                                         (short)0, c0, false, false);
            c1 = __builtin_amdgcn_wmma_f32_16x16x32_bf16(false, a1.v, false, b.v,
                                                         (short)0, c1, false, false);
        }

        if (chunk == 7) {   // super-tile logits complete -> online logsumexp
            const int col = vbase + myColOff;
            if (col < V_DIM) {
                #pragma unroll
                for (int r = 0; r < 8; ++r) {
                    float l0 = c0[r];
                    float nm0 = fmaxf(mx0[r], l0);
                    sm0[r] = sm0[r] * __expf(mx0[r] - nm0) + __expf(l0 - nm0);
                    mx0[r] = nm0;
                    if (tg0[r] == col) tgtL[r + 8 * hi] = l0;

                    float l1 = c1[r];
                    float nm1 = fmaxf(mx1[r], l1);
                    sm1[r] = sm1[r] * __expf(mx1[r] - nm1) + __expf(l1 - nm1);
                    mx1[r] = nm1;
                    if (tg1[r] == col) tgtL[16 + r + 8 * hi] = l1;
                }
            }
            c0 = (v8f){};
            c1 = (v8f){};
        }
        __syncthreads();
    }

    // ---- reduce across the 16 lanes holding the same rows ---------------
    #pragma unroll
    for (int r = 0; r < 8; ++r) {
        #pragma unroll
        for (int d = 1; d < 16; d <<= 1) {
            float om = __shfl_xor(mx0[r], d, 32);
            float os = __shfl_xor(sm0[r], d, 32);
            lse_merge(mx0[r], sm0[r], om, os);
            om = __shfl_xor(mx1[r], d, 32);
            os = __shfl_xor(sm1[r], d, 32);
            lse_merge(mx1[r], sm1[r], om, os);
        }
    }
    if (n == 0) {
        #pragma unroll
        for (int r = 0; r < 8; ++r) {
            stM[wv * M_TILE + r + 8 * hi]      = mx0[r];
            stS[wv * M_TILE + r + 8 * hi]      = sm0[r];
            stM[wv * M_TILE + 16 + r + 8 * hi] = mx1[r];
            stS[wv * M_TILE + 16 + r + 8 * hi] = sm1[r];
        }
    }
    __syncthreads();

    if (tid < M_TILE) {
        float M = NEG_BIG, S = 0.0f;
        #pragma unroll
        for (int wvi = 0; wvi < 8; ++wvi)
            lse_merge(M, S, stM[wvi * M_TILE + tid], stS[wvi * M_TILE + tid]);
        float lse = M + __logf(S);

        int t = tgt[mbase + tid];
        bool valid = (t != IGNORE_IDX);
        float loss = valid ? (lse - tgtL[tid]) : 0.0f;
        float cnt  = valid ? 1.0f : 0.0f;
        #pragma unroll
        for (int d = 1; d < 32; d <<= 1) {
            loss += __shfl_xor(loss, d, 32);
            cnt  += __shfl_xor(cnt, d, 32);
        }
        if (tid == 0) {
            atomicAdd(ws + 0, loss);
            atomicAdd(ws + 1, cnt);
        }
    }
}

// ---- pre-pass: x -> bf16 --------------------------------------------------
__global__ __launch_bounds__(256) void OlmoLMHeadCE_cvt_x(
    const float* __restrict__ x, unsigned short* __restrict__ xb) {
    int i = (blockIdx.x * 256 + threadIdx.x) * 4;   // 8,388,608 elems total
    const float4 f = *(const float4*)(x + i);
    uint2 p;
    p.x = (unsigned)f32_to_bf16_rne(f.x) | ((unsigned)f32_to_bf16_rne(f.y) << 16);
    p.y = (unsigned)f32_to_bf16_rne(f.z) | ((unsigned)f32_to_bf16_rne(f.w) << 16);
    *(uint2*)(xb + i) = p;
}

// ---- pre-pass: W[k][v] f32 -> wT[v][k] bf16, zero-padded to V_PAD ---------
__global__ __launch_bounds__(256) void OlmoLMHeadCE_cvt_wT(
    const float* __restrict__ w, unsigned short* __restrict__ wT) {
    __shared__ unsigned short t[32][33];
    const int vb = blockIdx.x * 32;
    const int kb = blockIdx.y * 32;
    #pragma unroll
    for (int i = 0; i < 4; ++i) {
        int idx = i * 256 + threadIdx.x;
        int kk = idx >> 5, vv = idx & 31;
        int col = vb + vv;
        float val = (col < V_DIM) ? w[(size_t)(kb + kk) * V_DIM + col] : 0.0f;
        t[vv][kk] = f32_to_bf16_rne(val);
    }
    __syncthreads();
    #pragma unroll
    for (int i = 0; i < 4; ++i) {
        int idx = i * 256 + threadIdx.x;
        int vv = idx >> 5, kk = idx & 31;
        wT[(size_t)(vb + vv) * H_DIM + kb + kk] = t[vv][kk];
    }
}

__global__ void OlmoLMHeadCE_init(float* ws) { ws[0] = 0.0f; ws[1] = 0.0f; }

__global__ void OlmoLMHeadCE_finalize(const float* ws, float* out) {
    out[0] = ws[0] / fmaxf(ws[1], 1.0f);
}

extern "C" void kernel_launch(void* const* d_in, const int* in_sizes, int n_in,
                              void* d_out, int out_size, void* d_ws, size_t ws_size,
                              hipStream_t stream) {
    (void)in_sizes; (void)n_in; (void)out_size; (void)ws_size;
    const float* x = (const float*)d_in[0];
    const float* w = (const float*)d_in[1];
    const int*   t = (const int*)d_in[2];
    float* out = (float*)d_out;
    float* ws  = (float*)d_ws;
    unsigned short* xb = (unsigned short*)((char*)d_ws + WS_XB_OFF);
    unsigned short* wT = (unsigned short*)((char*)d_ws + WS_WT_OFF);

    (void)hipFuncSetAttribute((const void*)OlmoLMHeadCE_main,
                              hipFuncAttributeMaxDynamicSharedMemorySize, SMEM_TOTAL);

    OlmoLMHeadCE_init<<<1, 1, 0, stream>>>(ws);
    OlmoLMHeadCE_cvt_x<<<(N_ROWS * H_DIM) / 1024, 256, 0, stream>>>(x, xb);
    OlmoLMHeadCE_cvt_wT<<<dim3(V_PAD / 32, H_DIM / 32), 256, 0, stream>>>(w, wT);
    OlmoLMHeadCE_main<<<N_ROWS / M_TILE, 256, SMEM_TOTAL, stream>>>(xb, wT, t, ws);
    OlmoLMHeadCE_finalize<<<1, 1, 0, stream>>>(ws, out);
}
